// RoSE_68607807586868
// MI455X (gfx1250) — matrix operand order repaired and checked by reading, not compiled, not run on hardware
//
#include <hip/hip_runtime.h>

typedef __attribute__((ext_vector_type(16))) _Float16 v16h;
typedef __attribute__((ext_vector_type(8)))  float    v8f;

#define T_STEPS 8
#define BATCH   4
#define C_IN    5
#define HH      256
#define WW      320
#define CB      32
#define TAU_INV 0.5f
#define BN_EPS  1e-5f

// tiling
#define TR  8     // output rows per block
#define TC  16    // output cols per block
#define R1  10    // conv1/LIF1 region rows (TR + 2 halo for conv2)
#define C1  18    // conv1/LIF1 region cols
#define XH  12    // staged x rows (R1 + 2 halo for conv1)
#define XW  20    // staged x cols
#define K1  45    // conv1 K = 5*9
#define K1P 64    // padded to 2 WMMA k-steps
#define K2  288   // conv2 K = 32*9 = 9 WMMA k-steps exactly
#define NXT (T_STEPS*C_IN*XH*XW)         // 9600 staged x elements

// LDS layout (bytes)
#define OFF_X   0
#define SZ_X    (NXT*2)                  // 19200  f16 input tile
#define OFF_W1  (OFF_X + SZ_X)
#define SZ_W1   (K1P*CB*2)               // 4096   f16 folded conv1 weights
#define OFF_W2  (OFF_W1 + SZ_W1)
#define SZ_W2   (K2*16*2)                // 9216   f16 folded conv2 weights (N padded to 16)
#define OFF_B1  (OFF_W2 + SZ_W2)         // 32512
#define OFF_B2  (OFF_B1 + 128)
#define OFF_Y   (OFF_B2 + 64)            // 32704  f16 conv1-out / spikes; ALSO aliased as
                                         //        the f32 async staging panel (38400 B) early on
#define SZ_Y    (T_STEPS*R1*C1*CB*2)     // 92160
#define OFF_Z   (OFF_Y + SZ_Y)           // 124864 f32 conv2 pre-activation
#define SZ_Z    (T_STEPS*TR*TC*4)        // 4096
#define SMEM_BYTES (OFF_Z + SZ_Z)        // 128960 (< 320KB/WGP; dynamic LDS; ~2 WG/WGP)

// ---- CDNA5 async global->LDS path (guarded; falls back to plain loads) ----
#if defined(__has_builtin)
#if __has_builtin(__builtin_amdgcn_global_load_async_to_lds_b32)
#define USE_ASYNC 1
#endif
#endif
#ifndef USE_ASYNC
#define USE_ASYNC 0
#endif

#if USE_ASYNC
typedef __attribute__((address_space(1))) int* gint_p;   // global (AS1)
typedef __attribute__((address_space(3))) int* lint_p;   // LDS (AS3)
__device__ __forceinline__ gint_p as_global(const void* p) {
  return (gint_p)(unsigned long long)(uintptr_t)p;
}
__device__ __forceinline__ lint_p as_lds(void* p) {
  // generic LDS address truncates to the 32-bit LDS offset (aperture mapping)
  return (lint_p)(unsigned)(uintptr_t)p;
}
__device__ __forceinline__ void wait_async0() {
#if __has_builtin(__builtin_amdgcn_s_wait_asynccnt)
  __builtin_amdgcn_s_wait_asynccnt(0);
#else
  asm volatile("s_wait_asynccnt 0x0" ::: "memory");
#endif
}
#endif

// ---------------- prep: fold BN into f16 weight matrices in ws ----------------
__global__ void snn_prep(const float* __restrict__ w1, const float* __restrict__ b1,
                         const float* __restrict__ g1, const float* __restrict__ be1,
                         const float* __restrict__ m1, const float* __restrict__ v1,
                         const float* __restrict__ w2, const float* __restrict__ b2,
                         const float* __restrict__ g2, const float* __restrict__ be2,
                         const float* __restrict__ m2, const float* __restrict__ v2,
                         _Float16* __restrict__ w1f, _Float16* __restrict__ w2f,
                         float* __restrict__ b1f, float* __restrict__ b2f) {
  int tid = threadIdx.x;
  for (int idx = tid; idx < K1P * CB; idx += 256) {
    int k = idx >> 5, oc = idx & 31;
    float inv = g1[oc] * rsqrtf(v1[oc] + BN_EPS);
    float w = 0.f;
    if (k < K1) {
      int ic = k / 9, j = k - ic * 9;
      w = w1[(oc * C_IN + ic) * 9 + j] * inv;   // OIHW
    }
    w1f[k * CB + oc] = (_Float16)w;
  }
  for (int idx = tid; idx < K2 * 16; idx += 256) {
    int k = idx >> 4, n = idx & 15;
    float w = 0.f;
    if (n == 0) {
      float inv = g2[0] * rsqrtf(v2[0] + BN_EPS);
      w = w2[k] * inv;                          // [1,32,3,3] flat = ic*9 + j
    }
    w2f[k * 16 + n] = (_Float16)w;
  }
  if (tid < CB) {
    float inv = g1[tid] * rsqrtf(v1[tid] + BN_EPS);
    b1f[tid] = b1[tid] * inv + be1[tid] - m1[tid] * inv;
  }
  if (tid == 0) {
    float inv = g2[0] * rsqrtf(v2[0] + BN_EPS);
    b2f[0] = b2[0] * inv + be2[0] - m2[0] * inv;
  }
}

// ---------------- fused conv1+BN -> LIF -> conv2+BN -> LIF ----------------
__global__ void __launch_bounds__(256, 1)
snn_fused(const float* __restrict__ x,
          const _Float16* __restrict__ w1f, const _Float16* __restrict__ w2f,
          const float* __restrict__ b1f, const float* __restrict__ b2f,
          float* __restrict__ out) {
  extern __shared__ char smem[];
  _Float16* sX  = (_Float16*)(smem + OFF_X);
  _Float16* sW1 = (_Float16*)(smem + OFF_W1);
  _Float16* sW2 = (_Float16*)(smem + OFF_W2);
  float*    sB1 = (float*)   (smem + OFF_B1);
  float*    sB2 = (float*)   (smem + OFF_B2);
  _Float16* sY  = (_Float16*)(smem + OFF_Y);
  float*    sZ  = (float*)   (smem + OFF_Z);

  const int tid  = threadIdx.x;
  const int lane = tid & 31;
  const int wv   = tid >> 5;            // wave id == timestep t (8 waves, T=8)
  const int nloc = lane & 15;           // WMMA N column (B,C,D) / M row (A)
  const int hi8  = (lane & 16) ? 8 : 0; // A: K bump for upper lanes; D: M bump
  const int kOff = lane & 16;           // unused; B fragments also use hi8 as K bump

  const int col0 = blockIdx.x * TC;
  const int row0 = blockIdx.y * TR;
  const int b    = blockIdx.z;

  // hint HBM: prefetch this wave's timestep plane near the tile
  __builtin_prefetch(x + (((size_t)(wv * BATCH + b)) * C_IN) * HH * WW
                       + (size_t)row0 * WW + col0, 0, 1);

  // ---- stage weights ----
  for (int i = tid; i < K1P * CB; i += 256) sW1[i] = w1f[i];
  for (int i = tid; i < K2 * 16;  i += 256) sW2[i] = w2f[i];
  if (tid < CB) sB1[tid] = b1f[tid];
  if (tid == 0) sB2[0] = b2f[0];

#if USE_ASYNC
  // ---- stage x via ASYNC global->LDS (no VGPR bounce), f32 panel aliases sY ----
  {
    float* sXF = (float*)(smem + OFF_Y);     // 38400 B staging panel
    for (int i = tid; i < NXT; i += 256) {
      int cc = i % XW;
      int tmp = i / XW;
      int rr = tmp % XH; tmp /= XH;
      int c  = tmp % C_IN;
      int t  = tmp / C_IN;
      int gr = row0 - 2 + rr, gc = col0 - 2 + cc;
      if (gr >= 0 && gr < HH && gc >= 0 && gc < WW) {   // EXEC-masked per lane
        const float* gp = x + ((((size_t)t * BATCH + b) * C_IN + c) * HH + gr) * WW + gc;
        __builtin_amdgcn_global_load_async_to_lds_b32(as_global(gp), as_lds(sXF + i), 0, 0);
      }
    }
    wait_async0();          // ASYNCcnt -> 0: LDS writes of this wave complete
    __syncthreads();        // make every wave's staged data visible
    // f32 -> f16 convert (out-of-image halo becomes exact 0 = SAME padding)
    for (int i = tid; i < NXT; i += 256) {
      int cc = i % XW;
      int tmp = i / XW;
      int rr = tmp % XH; tmp /= XH;
      int gr = row0 - 2 + rr, gc = col0 - 2 + cc;
      bool in = (gr >= 0 && gr < HH && gc >= 0 && gc < WW);
      sX[i] = (_Float16)(in ? sXF[i] : 0.f);
    }
  }
#else
  // ---- fallback: direct f32 load -> f16 LDS store ----
  for (int i = tid; i < NXT; i += 256) {
    int cc = i % XW;
    int tmp = i / XW;
    int rr = tmp % XH; tmp /= XH;
    int c  = tmp % C_IN;
    int t  = tmp / C_IN;
    int gr = row0 - 2 + rr, gc = col0 - 2 + cc;
    float v = 0.f;
    if (gr >= 0 && gr < HH && gc >= 0 && gc < WW)
      v = x[((((size_t)t * BATCH + b) * C_IN + c) * HH + gr) * WW + gc];
    sX[i] = (_Float16)v;
  }
#endif
  __syncthreads();          // sX ready; sY (staging panel) now dead until conv1 writes it

  // ---- conv1 B fragments: B[32x16], col=nloc(+16*nt), K=hi8+(h<8?h:h+8) (+32*ks) ----
  v16h bw1[2][2];
#pragma unroll
  for (int nt = 0; nt < 2; ++nt)
#pragma unroll
    for (int ks = 0; ks < 2; ++ks)
#pragma unroll
      for (int h = 0; h < 16; ++h)
        bw1[nt][ks][h] = sW1[(ks * 32 + hi8 + (h < 8 ? h : h + 8)) * CB + nt * 16 + nloc];

  // ---- conv1 over 10x18 halo region; M = 16 cols of one row ----
#pragma unroll 1
  for (int r1 = 0; r1 < R1; ++r1) {
#pragma unroll 1
    for (int cg = 0; cg < 2; ++cg) {          // col groups 0..15 and 2..17
      const int cBase = cg * 2;
      v16h a0, a1;
#pragma unroll
      for (int h = 0; h < 16; ++h) {
        int kk = hi8 + (h < 8 ? h : h + 8);   // A-matrix K mapping (ISA 7.12.2)
        _Float16 e0 = (_Float16)0.f, e1 = (_Float16)0.f;
        if (kk < K1) {                         // k-step 0
          int ic = kk / 9, j = kk - ic * 9, ky = j / 3, kx = j - ky * 3;
          e0 = sX[((wv * C_IN + ic) * XH + r1 + ky) * XW + cBase + nloc + kx];
        }
        int k1 = kk + 32;                      // k-step 1 (45..63 zero-padded)
        if (k1 < K1) {
          int ic = k1 / 9, j = k1 - ic * 9, ky = j / 3, kx = j - ky * 3;
          e1 = sX[((wv * C_IN + ic) * XH + r1 + ky) * XW + cBase + nloc + kx];
        }
        a0[h] = e0; a1[h] = e1;
      }
#pragma unroll
      for (int nt = 0; nt < 2; ++nt) {
        v8f acc = {};
        acc = __builtin_amdgcn_wmma_f32_16x16x32_f16(false, a0, false, bw1[nt][0],
                                                     (short)0, acc, false, false);
        acc = __builtin_amdgcn_wmma_f32_16x16x32_f16(false, a1, false, bw1[nt][1],
                                                     (short)0, acc, false, false);
        int oc = nt * 16 + nloc;
        float bias = sB1[oc];
#pragma unroll
        for (int i = 0; i < 8; ++i) {
          int m = hi8 + i;                    // D: M = vgpr + 8*(lane>=16)
          if (cg == 0 || m >= 14)             // cg1 contributes only cols 16,17
            sY[((wv * R1 + r1) * C1 + cBase + m) * CB + oc] = (_Float16)(acc[i] + bias);
        }
      }
    }
  }
  __syncthreads();

  // ---- LIF1: per-(pixel,oc) scan over T, spikes overwrite sY in place ----
  for (int ch = tid; ch < R1 * C1 * CB; ch += 256) {
    int oc = ch & 31;
    int pix = ch >> 5;
    int c1 = pix % C1, r1 = pix / C1;
    int gy = row0 - 1 + r1, gx = col0 - 1 + c1;
    bool in = (gy >= 0 && gy < HH && gx >= 0 && gx < WW); // SAME-pad spikes = 0
    float v = 0.f;
    for (int t = 0; t < T_STEPS; ++t) {
      int idx = ((t * R1 + r1) * C1 + c1) * CB + oc;
      float y = (float)sY[idx];
      v += (y - v) * TAU_INV;
      float s = (in && v >= 1.0f) ? 1.0f : 0.f;
      sY[idx] = (_Float16)s;
      v *= (1.0f - s);
    }
  }
  __syncthreads();

  // ---- conv2 B fragments: K2 = 288 = 9 k-steps, N padded to 16 (col 0 live) ----
  v16h bw2[9];
#pragma unroll
  for (int ks = 0; ks < 9; ++ks)
#pragma unroll
    for (int h = 0; h < 16; ++h)
      bw2[ks][h] = sW2[(ks * 32 + hi8 + (h < 8 ? h : h + 8)) * 16 + nloc];

#pragma unroll 1
  for (int r = 0; r < TR; ++r) {
    v8f acc = {};
#pragma unroll
    for (int ks = 0; ks < 9; ++ks) {
      v16h a;
#pragma unroll
      for (int h = 0; h < 16; ++h) {
        int k = ks * 32 + hi8 + (h < 8 ? h : h + 8);   // always < 288
        int ic = k / 9, j = k - ic * 9, ky = j / 3, kx = j - ky * 3;
        a[h] = sY[((wv * R1 + r + ky) * C1 + nloc + kx) * CB + ic];
      }
      acc = __builtin_amdgcn_wmma_f32_16x16x32_f16(false, a, false, bw2[ks],
                                                   (short)0, acc, false, false);
    }
    if (nloc == 0) {                         // only N column 0 is the real channel
#pragma unroll
      for (int i = 0; i < 8; ++i)
        sZ[(wv * TR + r) * TC + hi8 + i] = acc[i] + sB2[0];
    }
  }
  __syncthreads();

  // ---- LIF2 + write spikes to HBM ----
  if (tid < TR * TC) {
    int c = tid & 15, r = tid >> 4;
    int gy = row0 + r, gx = col0 + c;
    float v = 0.f;
    for (int t = 0; t < T_STEPS; ++t) {
      float z = sZ[(t * TR + r) * TC + c];
      v += (z - v) * TAU_INV;
      float s = (v >= 1.0f) ? 1.0f : 0.f;
      out[(((size_t)t * BATCH + b) * HH + gy) * WW + gx] = s;
      v *= (1.0f - s);
    }
  }
}

extern "C" void kernel_launch(void* const* d_in, const int* in_sizes, int n_in,
                              void* d_out, int out_size, void* d_ws, size_t ws_size,
                              hipStream_t stream) {
  (void)in_sizes; (void)n_in; (void)out_size; (void)ws_size;
  const float* x   = (const float*)d_in[0];
  const float* w1  = (const float*)d_in[1];
  const float* b1  = (const float*)d_in[2];
  const float* g1  = (const float*)d_in[3];
  const float* be1 = (const float*)d_in[4];
  const float* m1  = (const float*)d_in[5];
  const float* v1  = (const float*)d_in[6];
  const float* w2  = (const float*)d_in[7];
  const float* b2  = (const float*)d_in[8];
  const float* g2  = (const float*)d_in[9];
  const float* be2 = (const float*)d_in[10];
  const float* m2  = (const float*)d_in[11];
  const float* v2  = (const float*)d_in[12];

  char* ws = (char*)d_ws;
  _Float16* w1f = (_Float16*)(ws);            // 4096 B
  _Float16* w2f = (_Float16*)(ws + 4096);     // 9216 B
  float*    b1f = (float*)   (ws + 13312);    // 128 B
  float*    b2f = (float*)   (ws + 13440);    // 4 B

  snn_prep<<<1, 256, 0, stream>>>(w1, b1, g1, be1, m1, v1,
                                  w2, b2, g2, be2, m2, v2,
                                  w1f, w2f, b1f, b2f);

  dim3 grid(WW / TC, HH / TR, BATCH);         // 20 x 32 x 4 = 2560 workgroups
  snn_fused<<<grid, 256, SMEM_BYTES, stream>>>(x, w1f, w2f, b1f, b2f, (float*)d_out);
}
